// GraphSageLinkPredictor_43130061586688
// MI455X (gfx1250) — matrix operand order repaired and checked by reading, not compiled, not run on hardware
//
#include <hip/hip_runtime.h>
#include <hip/hip_bf16.h>

typedef __attribute__((ext_vector_type(16))) _Float16 v16h;
typedef __attribute__((ext_vector_type(8)))  float    v8f;

#define D 128

// ---------------- zero (grid-stride float4) ----------------
__global__ void zero_kernel(float4* __restrict__ p, long long n4) {
    long long i = (long long)blockIdx.x * blockDim.x + threadIdx.x;
    long long stride = (long long)gridDim.x * blockDim.x;
    float4 z; z.x = z.y = z.z = z.w = 0.0f;
    for (; i < n4; i += stride) p[i] = z;
}

// ---------------- degree count ----------------
__global__ void degree_kernel(const int* __restrict__ dst, float* __restrict__ cnt, int E) {
    int e = blockIdx.x * blockDim.x + threadIdx.x;
    if (e < E) atomicAdd(&cnt[dst[e]], 1.0f);
}

// ---------------- weights fp32 -> fp16 ----------------
__global__ void wcvt_kernel(const float* __restrict__ wl0, const float* __restrict__ wr0,
                            const float* __restrict__ wl1, const float* __restrict__ wr1,
                            _Float16* __restrict__ w16) {
    int i = blockIdx.x * blockDim.x + threadIdx.x;
    if (i < D * D) {
        w16[0 * D * D + i] = (_Float16)wl0[i];
        w16[1 * D * D + i] = (_Float16)wr0[i];
        w16[2 * D * D + i] = (_Float16)wl1[i];
        w16[3 * D * D + i] = (_Float16)wr1[i];
    }
}

// ---------------- edge scatter: agg[dst] += h[(x?)src] ----------------
// one thread per (edge, 4-feature chunk): 32 threads cover the 128 features of one edge
__global__ void scatter_kernel(const float* __restrict__ h,
                               const int* __restrict__ src,
                               const int* __restrict__ dst,
                               const int* __restrict__ xmap,   // NULL for layer 1
                               float* __restrict__ agg, int E) {
    long long idx = (long long)blockIdx.x * blockDim.x + threadIdx.x;
    int e = (int)(idx >> 5);
    if (e >= E) return;
    int c = ((int)idx & 31) << 2;
    int s = src[e];
    if (xmap) s = xmap[s];
    const float4 val = *(const float4*)(h + (long long)s * D + c);
    float* a = agg + (long long)dst[e] * D + c;
    atomicAdd(a + 0, val.x);
    atomicAdd(a + 1, val.y);
    atomicAdd(a + 2, val.z);
    atomicAdd(a + 3, val.w);
}

// ---------------- fused SAGE layer:  relu( (agg/cnt)@Wl^T + h@Wr^T + b ) ----------------
// 256 threads = 8 wave32 waves. Workgroup owns a 128-row tile; wave w owns rows [w*16, w*16+16).
// K = 256 (128 mean-part + 128 h-part), 8 chunks of v_wmma_f32_16x16x32_f16, 8 N-tiles.
// Per chunk: batch all 16 ds_load_b128 (8 B fragments), convert the double-buffered raw
// A data, prefetch next chunk's raw A (4 distinct float4 regs), then 8 WMMAs back-to-back.
__global__ __launch_bounds__(256)
void sage_gemm_kernel(const float* __restrict__ agg, const float* __restrict__ cnt,
                      const float* __restrict__ h_in, const int* __restrict__ xmap,
                      const _Float16* __restrict__ w16,   // [Wl(128x128) | Wr(128x128)] row-major f16
                      const float* __restrict__ bias,
                      float* __restrict__ out, int N) {
    __shared__ alignas(32) _Float16 sW[2 * D * D];   // 64 KB

    // cooperative 64KB weight stage (uint4 = 16B per thread per iter)
    {
        const uint4* g = (const uint4*)w16;
        uint4* s4 = (uint4*)sW;
        for (int i = threadIdx.x; i < (2 * D * D) / 8; i += 256) s4[i] = g[i];
    }
    __syncthreads();

    const int wave = threadIdx.x >> 5;
    const int lane = threadIdx.x & 31;
    const int l16  = lane & 15;
    const int hi   = lane >> 4;                     // 0: lanes 0-15, 1: lanes 16-31

    const int rowBase = blockIdx.x * 128 + wave * 16;
    int m  = rowBase + l16;                         // the row this lane supplies A data for
    int mc = m < N ? m : (N - 1);                   // clamp OOB loads; stores guarded below

    const float inv = 1.0f / fmaxf(cnt[mc], 1.0f);
    const float* aggRow = agg + (long long)mc * D;
    const int hrow = xmap ? xmap[mc] : mc;          // layer0: h = emb[x]
    const float* hRow = h_in + (long long)hrow * D;

    // this lane's B-fragment base inside LDS: weight row j = l16 (plus nt*16 per tile),
    // 16 contiguous halves starting at K = kc + (hi ? 16 : 0)
    const _Float16* bBase = sW + (long long)l16 * D + (hi ? 16 : 0);
    const int koff = hi ? 16 : 0;

    v8f acc[8];
#pragma unroll
    for (int i = 0; i < 8; ++i) acc[i] = (v8f){};

    // double-buffered raw A data: 4 distinct float4 regs so the 4 global_load_b128
    // stay independently in flight (no dest-register reuse serialization)
    float4 f0 = *(const float4*)(aggRow + koff);
    float4 f1 = *(const float4*)(aggRow + koff + 4);
    float4 f2 = *(const float4*)(aggRow + koff + 8);
    float4 f3 = *(const float4*)(aggRow + koff + 12);

#pragma unroll
    for (int chunk = 0; chunk < 8; ++chunk) {
        const bool meanPart = chunk < 4;
        const int  kc       = (chunk & 3) * 32;

        // ---- issue ALL B-fragment LDS loads for this chunk (16x ds_load_b128) ----
        v16h b[8];
        const _Float16* wb = bBase + (meanPart ? 0 : D * D) + kc;
#pragma unroll
        for (int nt = 0; nt < 8; ++nt) {
            b[nt] = *(const v16h*)(wb + nt * 16 * D);   // rows j = nt*16 + l16
        }

        // ---- convert the prefetched raw A data to the f16 fragment ----
        // A fragment in this code's contiguous k order (B matches): lanes 0-15 hold K = kc+{0..15},
        // lanes 16-31 hold K = kc+{16..31}; vector elems 2v/2v+1 -> VGPR v.
        const float scale = meanPart ? inv : 1.0f;
        v16h a;
        a[0]  = (_Float16)(f0.x * scale); a[1]  = (_Float16)(f0.y * scale);
        a[2]  = (_Float16)(f0.z * scale); a[3]  = (_Float16)(f0.w * scale);
        a[4]  = (_Float16)(f1.x * scale); a[5]  = (_Float16)(f1.y * scale);
        a[6]  = (_Float16)(f1.z * scale); a[7]  = (_Float16)(f1.w * scale);
        a[8]  = (_Float16)(f2.x * scale); a[9]  = (_Float16)(f2.y * scale);
        a[10] = (_Float16)(f2.z * scale); a[11] = (_Float16)(f2.w * scale);
        a[12] = (_Float16)(f3.x * scale); a[13] = (_Float16)(f3.y * scale);
        a[14] = (_Float16)(f3.z * scale); a[15] = (_Float16)(f3.w * scale);

        // ---- prefetch next chunk's raw A while the WMMAs execute ----
        if (chunk < 7) {
            const bool nmean = (chunk + 1) < 4;
            const int  nkc   = ((chunk + 1) & 3) * 32;
            const float* r = (nmean ? aggRow : hRow) + nkc + koff;
            f0 = *(const float4*)(r);
            f1 = *(const float4*)(r + 4);
            f2 = *(const float4*)(r + 8);
            f3 = *(const float4*)(r + 12);
        }

        // ---- 8 back-to-back WMMAs on the XDL pipe ----
#pragma unroll
        for (int nt = 0; nt < 8; ++nt) {
            acc[nt] = __builtin_amdgcn_wmma_f32_16x16x32_f16(
                false, a, false, b[nt], (short)0, acc[nt], false, false);
        }
    }

    // C layout: lane<16 -> rows rowBase+0..7 (VGPR v = row v), lane>=16 -> rows rowBase+8..15
#pragma unroll
    for (int nt = 0; nt < 8; ++nt) {
        const int j = nt * 16 + l16;
        const float bj = bias[j];
#pragma unroll
        for (int v = 0; v < 8; ++v) {
            const int row = rowBase + v + hi * 8;
            if (row < N) {
                out[(long long)row * D + j] = fmaxf(acc[nt][v] + bj, 0.0f);
            }
        }
    }
}

// ---------------- pair dot: one wave32 per pair ----------------
__global__ void dot_kernel(const float* __restrict__ h, const int* __restrict__ pairs,
                           float* __restrict__ out, int P) {
    int wid  = (int)(((long long)blockIdx.x * blockDim.x + threadIdx.x) >> 5);
    int lane = threadIdx.x & 31;
    if (wid >= P) return;
    const int u = pairs[2 * wid + 0];
    const int v = pairs[2 * wid + 1];
    const float4 a = *(const float4*)(h + (long long)u * D + lane * 4);
    const float4 b = *(const float4*)(h + (long long)v * D + lane * 4);
    float s = a.x * b.x + a.y * b.y + a.z * b.z + a.w * b.w;
#pragma unroll
    for (int off = 16; off > 0; off >>= 1) s += __shfl_xor(s, off, 32);
    if (lane == 0) out[wid] = s;
}

extern "C" void kernel_launch(void* const* d_in, const int* in_sizes, int n_in,
                              void* d_out, int out_size, void* d_ws, size_t ws_size,
                              hipStream_t stream) {
    const int*   x     = (const int*)  d_in[0];
    const int*   edges = (const int*)  d_in[1];   // [2,E]: src then dst
    const int*   pairs = (const int*)  d_in[2];   // [P,2]
    const float* emb   = (const float*)d_in[3];
    const float* wl0   = (const float*)d_in[4];
    const float* bl0   = (const float*)d_in[5];
    const float* wr0   = (const float*)d_in[6];
    const float* wl1   = (const float*)d_in[7];
    const float* bl1   = (const float*)d_in[8];
    const float* wr1   = (const float*)d_in[9];
    float* outp = (float*)d_out;

    const int N = in_sizes[3] / D;     // 100000
    const int E = in_sizes[1] / 2;     // 1.6M
    const int P = in_sizes[2] / 2;     // 1M
    const int* esrc = edges;
    const int* edst = edges + E;

    // workspace layout (bytes, 256B aligned)
    char* ws = (char*)d_ws;
    const long long cntBytes = (((long long)N * 4 + 255) / 256) * 256;           // 400128
    const long long matBytes = (long long)N * D * 4;                             // 51.2 MB
    float*    cnt = (float*)ws;
    float*    agg = (float*)(ws + cntBytes);                 // also holds h2 after layer 1
    float*    h1  = (float*)(ws + cntBytes + matBytes);
    _Float16* w16 = (_Float16*)(ws + cntBytes + 2 * matBytes);

    const int Z = 256;
    // 0) zero cnt+agg in one pass (regions are contiguous & 16B multiples)
    long long n4 = (cntBytes + matBytes) / 16;
    zero_kernel<<<2048, Z, 0, stream>>>((float4*)ws, n4);
    // weights to f16 once per call
    wcvt_kernel<<<(D * D + Z - 1) / Z, Z, 0, stream>>>(wl0, wr0, wl1, wr1, w16);
    // 1) degrees
    degree_kernel<<<(E + Z - 1) / Z, Z, 0, stream>>>(edst, cnt, E);
    // 2) layer 0: scatter emb[x[src]] -> agg, then fused WMMA layer to h1
    {
        long long tthreads = (long long)E * 32;
        scatter_kernel<<<(unsigned)((tthreads + Z - 1) / Z), Z, 0, stream>>>(emb, esrc, edst, x, agg, E);
        sage_gemm_kernel<<<(N + 127) / 128, Z, 0, stream>>>(agg, cnt, emb, x, w16, bl0, h1, N);
    }
    // 3) layer 1: re-zero agg, scatter h1[src] -> agg, fused layer written in-place over agg (= h2)
    zero_kernel<<<2048, Z, 0, stream>>>((float4*)agg, matBytes / 16);
    {
        long long tthreads = (long long)E * 32;
        scatter_kernel<<<(unsigned)((tthreads + Z - 1) / Z), Z, 0, stream>>>(h1, esrc, edst, nullptr, agg, E);
        sage_gemm_kernel<<<(N + 127) / 128, Z, 0, stream>>>(agg, cnt, h1, nullptr, w16 + 2 * D * D, bl1, agg, N);
    }
    // 4) pair dot products from h2 (= agg buffer)
    {
        long long tthreads = (long long)P * 32;
        dot_kernel<<<(unsigned)((tthreads + Z - 1) / Z), Z, 0, stream>>>(agg, pairs, outp, P);
    }
}
